// PowerFlowGINE_61177514164374
// MI455X (gfx1250) — matrix operand order repaired and checked by reading, not compiled, not run on hardware
//
#include <hip/hip_runtime.h>
#include <hip/hip_bf16.h>

// ---------------------------------------------------------------------------
// Types for CDNA5 WMMA (wave32): V_WMMA_F32_16X16X32_BF16
// ---------------------------------------------------------------------------
typedef __attribute__((ext_vector_type(16))) __bf16 v16bf;
typedef __attribute__((ext_vector_type(8)))  float  v8f;

union FragBF {
    uint4 q[2];   // 8 dwords = 16 bf16
    v16bf v;
};

// Native hardware converts (v_cvt_pk_bf16_f32 / v_cvt_bf16_f32) instead of
// software round-to-nearest-even bit twiddling.
__device__ __forceinline__ unsigned short f2bf(float f) {
    union { __bf16 h; unsigned short u; } x;
    x.h = (__bf16)f;
    return x.u;
}
__device__ __forceinline__ unsigned int pack2bf(float a, float b) {
    union { __bf16 h[2]; unsigned int u; } x;
    x.h[0] = (__bf16)a;
    x.h[1] = (__bf16)b;
    return x.u;
}

// ---------------------------------------------------------------------------
// Weight prep: f32 [nmat][256][256] row-major -> bf16 transposed [nmat][col][k]
// ---------------------------------------------------------------------------
__global__ void prep_weights(const float* __restrict__ src,
                             unsigned short* __restrict__ dst, int nmat) {
    int idx = blockIdx.x * blockDim.x + threadIdx.x;
    int total = nmat << 16;
    if (idx >= total) return;
    int mat = idx >> 16;
    int rem = idx & 65535;
    int col = rem >> 8;
    int k   = rem & 255;
    dst[idx] = f2bf(src[(mat << 16) + (k << 8) + col]);
}

// Fold bias + eval-mode BN into per-column affine: y = x*s + t (then ReLU)
__global__ void prep_bn(const float* __restrict__ b, const float* __restrict__ g,
                        const float* __restrict__ beta, const float* __restrict__ m,
                        const float* __restrict__ v, float* __restrict__ s,
                        float* __restrict__ t, int L) {
    int i = blockIdx.x * blockDim.x + threadIdx.x;
    if (i >= L) return;
    float sc = g[i] * rsqrtf(v[i] + 1e-5f);
    s[i] = sc;
    t[i] = (b[i] - m[i]) * sc + beta[i];
}

// ---------------------------------------------------------------------------
// Edge-attr standardization stats: per-column sum & sumsq over E rows (9 cols)
// ---------------------------------------------------------------------------
__global__ void zero_stats(float* stats) {
    if (threadIdx.x < 18) stats[threadIdx.x] = 0.f;
}

__global__ void edge_stats(const float* __restrict__ ea, float* stats, int E) {
    __shared__ float red[256];
    float acc[18];
#pragma unroll
    for (int j = 0; j < 18; j++) acc[j] = 0.f;
    for (int i = blockIdx.x * 256 + threadIdx.x; i < E; i += gridDim.x * 256) {
#pragma unroll
        for (int j = 0; j < 9; j++) {
            float v = ea[(size_t)i * 9 + j];
            acc[j]     += v;
            acc[9 + j] += v * v;
        }
    }
#pragma unroll 1
    for (int j = 0; j < 18; j++) {
        red[threadIdx.x] = acc[j];
        __syncthreads();
        for (int off = 128; off > 0; off >>= 1) {
            if (threadIdx.x < off) red[threadIdx.x] += red[threadIdx.x + off];
            __syncthreads();
        }
        if (threadIdx.x == 0) unsafeAtomicAdd(&stats[j], red[0]);
        __syncthreads();
    }
}

// ---------------------------------------------------------------------------
// Node lift: h = relu(x @ W[10,256] + b); also init running JK max
// ---------------------------------------------------------------------------
__global__ __launch_bounds__(256) void lift_kernel(
    const float* __restrict__ x, const float* __restrict__ W,
    const float* __restrict__ b, float* __restrict__ h,
    float* __restrict__ hmax, int N) {
    __shared__ float sx[10];
    int nid = blockIdx.x;
    int c = threadIdx.x;
    if (c < 10) sx[c] = x[(size_t)nid * 10 + c];
    __syncthreads();
    float acc = b[c];
#pragma unroll
    for (int i = 0; i < 10; i++) acc = fmaf(sx[i], W[(i << 8) + c], acc);
    float v = fmaxf(acc, 0.f);
    size_t idx = ((size_t)nid << 8) + c;
    h[idx] = v;
    hmax[idx] = v;
}

// Edge lift: e = relu(((ea - mu)/(sd+1e-8)) @ W[9,256] + b)
__global__ __launch_bounds__(256) void elift_kernel(
    const float* __restrict__ ea, const float* __restrict__ W,
    const float* __restrict__ b, const float* __restrict__ stats,
    float* __restrict__ e, int E) {
    __shared__ float sx[9];
    int eid = blockIdx.x;
    int c = threadIdx.x;
    if (c < 9) {
        float invE = 1.f / (float)E;
        float mu = stats[c] * invE;
        float var = stats[9 + c] * invE - mu * mu;
        float sd = sqrtf(fmaxf(var, 0.f));
        sx[c] = (ea[(size_t)eid * 9 + c] - mu) / (sd + 1e-8f);
    }
    __syncthreads();
    float acc = b[c];
#pragma unroll
    for (int i = 0; i < 9; i++) acc = fmaf(sx[i], W[(i << 8) + c], acc);
    e[((size_t)eid << 8) + c] = fmaxf(acc, 0.f);
}

// agg = h  (vectorized copy; agg then accumulates edge messages -> z = h + sum)
__global__ void copy4_kernel(const float4* __restrict__ src,
                             float4* __restrict__ dst, size_t n4) {
    size_t i = (size_t)blockIdx.x * blockDim.x + threadIdx.x;
    if (i < n4) dst[i] = src[i];
}

// msg = relu(h[src] + e); agg[dst] += msg   (native f32 global atomics)
__global__ __launch_bounds__(256) void scatter_kernel(
    const float* __restrict__ h, const float* __restrict__ e,
    const int* __restrict__ eidx, float* __restrict__ agg, int E) {
    int eid = blockIdx.x;
    int c = threadIdx.x;
    int s = eidx[eid];
    int d = eidx[E + eid];
    float v = h[((size_t)s << 8) + c] + e[((size_t)eid << 8) + c];
    v = fmaxf(v, 0.f);
    unsafeAtomicAdd(&agg[((size_t)d << 8) + c], v);
}

// ---------------------------------------------------------------------------
// WMMA GEMM: out[nrows,256] = relu( (A[nrows,256] @ B[256,256]) * scale + shift )
// B supplied transposed as bf16  Bt[col][k]  (L2-resident, 128KB).
// Block = 64 rows x 256 cols; 8 waves as 2(M) x 4(N); wave = 2x4 tiles of 16x16.
// A tile (64x32) double-buffered in LDS as bf16: next chunk's global_load_b128s
// issue before the 8 WMMAs, one barrier per K-step. Optional fused JK max.
// ---------------------------------------------------------------------------
__device__ __forceinline__ uint4 stage_loadA(const float* __restrict__ arow, int koff) {
    float4 f0 = {0.f, 0.f, 0.f, 0.f};
    float4 f1 = {0.f, 0.f, 0.f, 0.f};
    if (arow) {
        f0 = *(const float4*)(arow + koff);
        f1 = *(const float4*)(arow + koff + 4);
    }
    uint4 pk;
    pk.x = pack2bf(f0.x, f0.y);
    pk.y = pack2bf(f0.z, f0.w);
    pk.z = pack2bf(f1.x, f1.y);
    pk.w = pack2bf(f1.z, f1.w);
    return pk;
}

template <bool FUSE_MAX>
__global__ __launch_bounds__(256) void gemm256(
    const float* __restrict__ A, const unsigned short* __restrict__ Bt,
    const float* __restrict__ scale, const float* __restrict__ shift,
    float* __restrict__ out, float* __restrict__ hmax, int nrows) {
    __shared__ __align__(16) unsigned short Abuf[2][64 * 32];

    const int tid   = threadIdx.x;
    const int lane  = tid & 31;
    const int wid   = tid >> 5;
    const int waveM = wid >> 2;   // 0..1
    const int waveN = wid & 3;    // 0..3
    const int lmod  = lane & 15;
    const int lhalf = lane >> 4;
    const int rowBlock = blockIdx.x * 64;

    // staging assignment: thread -> (row, 8-wide k chunk)
    const int sr = tid >> 2;          // 0..63
    const int sk = (tid & 3) << 3;    // 0,8,16,24
    const int grow = rowBlock + sr;
    const float* arow = (grow < nrows) ? (A + ((size_t)grow << 8) + sk) : nullptr;

    v8f acc[2][4] = {};

    // prologue: stage K-chunk 0
    *(uint4*)&Abuf[0][(sr << 5) + sk] = stage_loadA(arow, 0);
    __syncthreads();

#pragma unroll
    for (int step = 0; step < 8; step++) {
        const int k0 = step * 32;
        const int cur = step & 1;

        // prefetch next A chunk into registers (overlaps with WMMAs below)
        uint4 nextA;
        if (step < 7) nextA = stage_loadA(arow, k0 + 32);

        // --- B fragments (32x16 per tile): 2 x b128 straight from L2 ---
        FragBF bf[4];
#pragma unroll
        for (int n = 0; n < 4; n++) {
            const unsigned short* bp =
                Bt + (((size_t)(waveN * 64 + n * 16 + lmod)) << 8) + k0 + (lhalf << 3);
            bf[n].q[0] = *(const uint4*)bp;
            bf[n].q[1] = *(const uint4*)(bp + 16);
        }
        // --- A fragments from LDS (16x32 per tile): 2 x ds_read_b128 ---
#pragma unroll
        for (int m = 0; m < 2; m++) {
            const int r = waveM * 32 + m * 16 + lmod;
            FragBF af;
            af.q[0] = *(const uint4*)&Abuf[cur][(r << 5) + (lhalf << 3)];
            af.q[1] = *(const uint4*)&Abuf[cur][(r << 5) + 16 + (lhalf << 3)];
#pragma unroll
            for (int n = 0; n < 4; n++) {
                acc[m][n] = __builtin_amdgcn_wmma_f32_16x16x32_bf16(
                    false, af.v, false, bf[n].v, (short)0, acc[m][n], false, false);
            }
        }

        if (step < 7) {
            // safe: last reads of buffer (cur^1) were in step-1, fenced by its barrier
            *(uint4*)&Abuf[cur ^ 1][(sr << 5) + sk] = nextA;
            __syncthreads();
        }
    }

    // --- epilogue: y = relu(x*s + t); optional hmax = max(hmax, y) ---
#pragma unroll
    for (int n = 0; n < 4; n++) {
        const int col = waveN * 64 + n * 16 + lmod;
        const float s = scale[col];
        const float t = shift[col];
#pragma unroll
        for (int m = 0; m < 2; m++) {
#pragma unroll
            for (int i = 0; i < 8; i++) {
                const int row = rowBlock + waveM * 32 + m * 16 + (lhalf << 3) + i;
                if (row < nrows) {
                    float y = fmaxf(fmaf(acc[m][n][i], s, t), 0.f);
                    const size_t idx = ((size_t)row << 8) + col;
                    out[idx] = y;
                    if (FUSE_MAX) hmax[idx] = fmaxf(hmax[idx], y);
                }
            }
        }
    }
}

// Final head: out[N,3] = z[N,256] @ pW3[256,3] + pb3
__global__ void head_kernel(const float* __restrict__ z, const float* __restrict__ W,
                            const float* __restrict__ b, float* __restrict__ out, int N) {
    int n = blockIdx.x * blockDim.x + threadIdx.x;
    if (n >= N) return;
    float a0 = b[0], a1 = b[1], a2 = b[2];
    const float* zr = z + ((size_t)n << 8);
#pragma unroll 4
    for (int k = 0; k < 256; k++) {
        float v = zr[k];
        a0 = fmaf(v, W[k * 3 + 0], a0);
        a1 = fmaf(v, W[k * 3 + 1], a1);
        a2 = fmaf(v, W[k * 3 + 2], a2);
    }
    out[(size_t)n * 3 + 0] = a0;
    out[(size_t)n * 3 + 1] = a1;
    out[(size_t)n * 3 + 2] = a2;
}

// ---------------------------------------------------------------------------
extern "C" void kernel_launch(void* const* d_in, const int* in_sizes, int n_in,
                              void* d_out, int out_size, void* d_ws, size_t ws_size,
                              hipStream_t stream) {
    const float* x         = (const float*)d_in[0];
    const float* edge_attr = (const float*)d_in[1];
    const int*   eidx      = (const int*)d_in[2];
    const float* lift_W    = (const float*)d_in[3];
    const float* lift_b    = (const float*)d_in[4];
    const float* elift_W   = (const float*)d_in[5];
    const float* elift_b   = (const float*)d_in[6];
    const float* conv_W1   = (const float*)d_in[7];
    const float* conv_b1   = (const float*)d_in[8];
    const float* conv_bng  = (const float*)d_in[9];
    const float* conv_bnb  = (const float*)d_in[10];
    const float* conv_bnm  = (const float*)d_in[11];
    const float* conv_bnv  = (const float*)d_in[12];
    const float* conv_W2   = (const float*)d_in[13];
    const float* conv_b2   = (const float*)d_in[14];
    const float* bng       = (const float*)d_in[15];
    const float* bnb       = (const float*)d_in[16];
    const float* bnm       = (const float*)d_in[17];
    const float* bnv       = (const float*)d_in[18];
    const float* pW1       = (const float*)d_in[19];
    const float* pb1       = (const float*)d_in[20];
    const float* pbn1g     = (const float*)d_in[21];
    const float* pbn1b     = (const float*)d_in[22];
    const float* pbn1m     = (const float*)d_in[23];
    const float* pbn1v     = (const float*)d_in[24];
    const float* pW2       = (const float*)d_in[25];
    const float* pb2       = (const float*)d_in[26];
    const float* pbn2g     = (const float*)d_in[27];
    const float* pbn2b     = (const float*)d_in[28];
    const float* pbn2m     = (const float*)d_in[29];
    const float* pbn2v     = (const float*)d_in[30];
    const float* pW3       = (const float*)d_in[31];
    const float* pb3       = (const float*)d_in[32];
    float* out = (float*)d_out;

    const int N = in_sizes[0] / 10;
    const int E = in_sizes[1] / 9;
    const int H = 256;
    const int T = in_sizes[7] / (H * H);

    const size_t NH = (size_t)N * H;
    const size_t EH = (size_t)E * H;

    // --- workspace layout ---
    unsigned short* Wt = (unsigned short*)d_ws;              // (2T+2) * 65536 bf16
    float* fws  = (float*)((char*)d_ws + (size_t)(2 * T + 2) * 65536 * 2);
    float* h    = fws;
    float* hmax = h + NH;
    float* agg  = hmax + NH;
    float* tmp  = agg + NH;
    float* e    = tmp + NH;
    float* s1   = e + EH;              float* sh1 = s1 + (size_t)T * H;
    float* s2   = sh1 + (size_t)T * H; float* sh2 = s2 + (size_t)T * H;
    float* ps1  = sh2 + (size_t)T * H;
    float* psh1 = ps1 + H;
    float* ps2  = psh1 + H;
    float* psh2 = ps2 + H;
    float* stats = psh2 + H;           // 18 floats

    // --- preprocessing ---
    zero_stats<<<1, 32, 0, stream>>>(stats);
    edge_stats<<<512, 256, 0, stream>>>(edge_attr, stats, E);

    prep_weights<<<(T * 65536 + 255) / 256, 256, 0, stream>>>(conv_W1, Wt, T);
    prep_weights<<<(T * 65536 + 255) / 256, 256, 0, stream>>>(conv_W2, Wt + (size_t)T * 65536, T);
    prep_weights<<<(65536 + 255) / 256, 256, 0, stream>>>(pW1, Wt + (size_t)(2 * T) * 65536, 1);
    prep_weights<<<(65536 + 255) / 256, 256, 0, stream>>>(pW2, Wt + (size_t)(2 * T + 1) * 65536, 1);

    prep_bn<<<(T * H + 255) / 256, 256, 0, stream>>>(conv_b1, conv_bng, conv_bnb, conv_bnm, conv_bnv, s1, sh1, T * H);
    prep_bn<<<(T * H + 255) / 256, 256, 0, stream>>>(conv_b2, bng, bnb, bnm, bnv, s2, sh2, T * H);
    prep_bn<<<1, 256, 0, stream>>>(pb1, pbn1g, pbn1b, pbn1m, pbn1v, ps1, psh1, H);
    prep_bn<<<1, 256, 0, stream>>>(pb2, pbn2g, pbn2b, pbn2m, pbn2v, ps2, psh2, H);

    // --- lifts ---
    lift_kernel<<<N, 256, 0, stream>>>(x, lift_W, lift_b, h, hmax, N);
    elift_kernel<<<E, 256, 0, stream>>>(edge_attr, elift_W, elift_b, stats, e, E);

    const int gemmBlocks = (N + 63) / 64;
    const int copyBlocks = (int)((NH / 4 + 255) / 256);

    // --- T GINE conv layers ---
    for (int t = 0; t < T; t++) {
        copy4_kernel<<<copyBlocks, 256, 0, stream>>>((const float4*)h, (float4*)agg, NH / 4);
        scatter_kernel<<<E, 256, 0, stream>>>(h, e, eidx, agg, E);
        gemm256<false><<<gemmBlocks, 256, 0, stream>>>(agg, Wt + (size_t)t * 65536,
                                                       s1 + (size_t)t * H, sh1 + (size_t)t * H,
                                                       tmp, nullptr, N);
        gemm256<true><<<gemmBlocks, 256, 0, stream>>>(tmp, Wt + (size_t)(T + t) * 65536,
                                                      s2 + (size_t)t * H, sh2 + (size_t)t * H,
                                                      h, hmax, N);
    }

    // --- post MLP (two BN+ReLU GEMMs then 256->3 head) ---
    gemm256<false><<<gemmBlocks, 256, 0, stream>>>(hmax, Wt + (size_t)(2 * T) * 65536, ps1, psh1, agg, nullptr, N);
    gemm256<false><<<gemmBlocks, 256, 0, stream>>>(agg, Wt + (size_t)(2 * T + 1) * 65536, ps2, psh2, tmp, nullptr, N);
    head_kernel<<<(N + 255) / 256, 256, 0, stream>>>(tmp, pW3, pb3, out, N);

    (void)n_in; (void)out_size; (void)ws_size;
}
